// MistralSelfAttention_62199716380849
// MI455X (gfx1250) — compile-verified, matrix operand order checked by reading, not run on hardware
//
#include <hip/hip_runtime.h>
#include <hip/hip_bf16.h>

// ---------------------------------------------------------------------------
// Problem constants (B,T,H,D) = (2,1024,32,128), KV=8, Q_GROUP=4
// ---------------------------------------------------------------------------
#define BB   2
#define TT   1024
#define HH   32
#define DD   128
#define KVH  8
#define MDIM (BB * TT)      // 2048 tokens
#define KDIM (HH * DD)      // 4096 model dim
#define NEGV (-1000000000.0f)

#if __has_builtin(__builtin_amdgcn_tensor_load_to_lds)
#define HAS_TDM 1
#else
#define HAS_TDM 0
#endif

typedef __attribute__((ext_vector_type(16))) __bf16 v16bf;
typedef __attribute__((ext_vector_type(8)))  float  v8f;
typedef __attribute__((ext_vector_type(4)))  unsigned int v4u;
typedef __attribute__((ext_vector_type(8)))  int    v8i;
typedef __attribute__((ext_vector_type(4)))  int    v4i;

union FragU { v16bf v; uint4 q[2]; };

static __device__ __forceinline__ unsigned short f32_to_bf16(float f) {
  union { float f; unsigned u; } c; c.f = f;
  unsigned u = c.u;
  u += 0x7FFFu + ((u >> 16) & 1u);           // round-to-nearest-even
  return (unsigned short)(u >> 16);
}

static __device__ __forceinline__ v8f wmma_bf16(v16bf a, v16bf b, v8f c) {
  return __builtin_amdgcn_wmma_f32_16x16x32_bf16(
      /*neg_a=*/false, a, /*neg_b=*/false, b,
      /*c_mod=*/(short)0, c, /*reuse_a=*/false, /*reuse_b=*/false);
}

// 16x32 bf16 fragment load (A layout: row striped within lane) or a 32x16
// B fragment from a row-major [N][K] source (lane holds row n, reads along k).
// Per the CDNA5 ISA layout: lane holds row (lane&15); K-pairs at
// kbase..kbase+7 and kbase+16..kbase+23 with kbase=(lane>>4)*8.
static __device__ __forceinline__ v16bf load_frag(const unsigned short* base,
                                                  int ld, int row, int col,
                                                  int lane) {
  int r  = row + (lane & 15);
  int kb = col + ((lane >> 4) << 3);
  const unsigned short* p = base + (size_t)r * ld + kb;
  FragU f;
  f.q[0] = *(const uint4*)(p);
  f.q[1] = *(const uint4*)(p + 16);
  return f.v;
}

// Issue one 2-D TDM tile load (global -> LDS). Descriptor per CDNA5 ISA §8:
// group0 = {count, lds_addr, global_addr, type=2}, group1 = {data_size=2B,
// tensor_dim0/1, tile_dim0/1, dim0_stride}. Wave-level op, TENSORcnt tracked.
static __device__ __forceinline__ void tdm_load_2d(unsigned lds_off,
                                                   const void* gptr,
                                                   unsigned dim0, unsigned dim1,
                                                   unsigned long long stride0) {
#if HAS_TDM
  unsigned long long ga = (unsigned long long)gptr;
  unsigned ga_lo = __builtin_amdgcn_readfirstlane((unsigned)ga);
  unsigned ga_hi = __builtin_amdgcn_readfirstlane((unsigned)(ga >> 32));
  unsigned lds_s = __builtin_amdgcn_readfirstlane(lds_off);
  v4u g0;
  g0[0] = 1u;                                  // count=1 (valid), user mode
  g0[1] = lds_s;                               // lds_addr
  g0[2] = ga_lo;                               // global_addr[31:0]
  g0[3] = (ga_hi & 0x01FFFFFFu) | (2u << 30);  // global_addr[56:32] | type=2
  v8i g1;
  g1[0] = (int)(1u << 16);                     // wg_mask=0, data_size=1 (2B)
  g1[1] = (int)((dim0 & 0xFFFFu) << 16);       // tensor_dim0[15:0]
  g1[2] = (int)((dim0 >> 16) | ((dim1 & 0xFFFFu) << 16));   // d0 hi | d1 lo
  g1[3] = (int)((dim1 >> 16) | (dim0 << 16));  // d1 hi | tile_dim0
  g1[4] = (int)(dim1 & 0xFFFFu);               // tile_dim1, tile_dim2=0
  g1[5] = (int)(unsigned)(stride0 & 0xFFFFFFFFu);           // dim0_stride lo
  g1[6] = (int)(unsigned)((stride0 >> 32) & 0xFFFFu);       // dim0_stride hi
  g1[7] = 0;
  v4i z4 = {0, 0, 0, 0};
#if __has_include(<hip/amd_detail/amd_gfx1250_TDM.h>)
  v8i z8 = {0, 0, 0, 0, 0, 0, 0, 0};
  __builtin_amdgcn_tensor_load_to_lds(g0, g1, z4, z4, z8, 0);   // clang-23 form
#else
  __builtin_amdgcn_tensor_load_to_lds(g0, g1, z4, z4, 0);       // ROCm 7.2 form
#endif
#endif
}

// ---------------------------------------------------------------------------
// Kernel 1: fp32 -> bf16 conversion (vectorized)
// ---------------------------------------------------------------------------
__global__ void __launch_bounds__(256) k_f32_to_bf16(const float* __restrict__ in,
                                                     unsigned short* __restrict__ out,
                                                     int n) {
  int i = (blockIdx.x * 256 + threadIdx.x) * 4;
  if (i >= n) return;
  float4 f = *(const float4*)(in + i);
  unsigned lo = (unsigned)f32_to_bf16(f.x) | ((unsigned)f32_to_bf16(f.y) << 16);
  unsigned hi = (unsigned)f32_to_bf16(f.z) | ((unsigned)f32_to_bf16(f.w) << 16);
  uint2 st; st.x = lo; st.y = hi;
  *(uint2*)(out + i) = st;
}

// ---------------------------------------------------------------------------
// Kernel 2: GEMM  C[M][N] = A[M][K] * B[N][K]^T   (bf16 in, fp32 out)
// Block = 256 threads (8 waves), block tile 128x128, K-step 64.
// A tile staged by the Tensor Data Mover (TENSORcnt), B tile by per-lane
// global_load_async_to_lds_b128 (ASYNCcnt). Each wave computes 32x64 with
// 8 v_wmma_f32_16x16x32_bf16 per 32-wide k-step.
// ---------------------------------------------------------------------------
__global__ void __launch_bounds__(256) k_gemm_bf16(const unsigned short* __restrict__ A,
                                                   const unsigned short* __restrict__ Bm,
                                                   float* __restrict__ C,
                                                   int M, int N, int K) {
  __shared__ unsigned short As[128][64];   // 16 KB, contiguous for TDM
  __shared__ unsigned short Bs[128][64];   // 16 KB

  const int tid  = threadIdx.x;
  const int lane = tid & 31;
  const int w    = tid >> 5;
  const int wm   = w & 3;          // 0..3 -> 32-row M sub-tile
  const int wn   = w >> 2;         // 0..1 -> 64-col N sub-tile
  const int m0   = blockIdx.y * 128;
  const int n0   = blockIdx.x * 128;

  v8f acc[2][4];
#pragma unroll
  for (int i = 0; i < 2; ++i)
#pragma unroll
    for (int j = 0; j < 4; ++j)
      acc[i][j] = (v8f){0.f,0.f,0.f,0.f,0.f,0.f,0.f,0.f};

  for (int kb = 0; kb < K; kb += 64) {
#if HAS_TDM
    // ---- A tile: one TDM descriptor, issued by wave 0 (128 rows x 64 cols) ----
    if (w == 0) {
      tdm_load_2d((unsigned)(unsigned long long)&As[0][0],
                  A + (size_t)m0 * K + kb, 64u, 128u, (unsigned long long)K);
    }
#else
    // ---- fallback: cooperative A-tile load through VGPRs ----
#pragma unroll
    for (int i = 0; i < 4; ++i) {
      int idx = tid + i * 256;                 // 0..1023
      int r = idx >> 3, c = (idx & 7) * 8;
      *(uint4*)&As[r][c] = *(const uint4*)(A + (size_t)(m0 + r) * K + kb + c);
    }
#endif
    // ---- B tile: per-lane async global->LDS copies (no VGPR data path) ----
#pragma unroll
    for (int i = 0; i < 4; ++i) {
      int idx = tid + i * 256;                 // 0..1023
      int r = idx >> 3, c = (idx & 7) * 8;
      unsigned lofs = (unsigned)(unsigned long long)&Bs[r][c];
      const unsigned short* gp = Bm + (size_t)(n0 + r) * K + kb + c;
      asm volatile("global_load_async_to_lds_b128 %0, %1, off"
                   :: "v"(lofs), "v"((unsigned long long)gp) : "memory");
    }
    if (kb + 64 < K)
      __builtin_prefetch(Bm + (size_t)(n0 + (tid >> 1)) * K + kb + 64, 0, 3);

    asm volatile("s_wait_asynccnt 0" ::: "memory");
    __builtin_amdgcn_s_wait_tensorcnt(0);
    __syncthreads();

#pragma unroll
    for (int kk = 0; kk < 64; kk += 32) {
      v16bf a0 = load_frag(&As[0][0], 64, wm * 32,      kk, lane);
      v16bf a1 = load_frag(&As[0][0], 64, wm * 32 + 16, kk, lane);
      v16bf b0 = load_frag(&Bs[0][0], 64, wn * 64,      kk, lane);
      v16bf b1 = load_frag(&Bs[0][0], 64, wn * 64 + 16, kk, lane);
      v16bf b2 = load_frag(&Bs[0][0], 64, wn * 64 + 32, kk, lane);
      v16bf b3 = load_frag(&Bs[0][0], 64, wn * 64 + 48, kk, lane);
      acc[0][0] = wmma_bf16(a0, b0, acc[0][0]);
      acc[0][1] = wmma_bf16(a0, b1, acc[0][1]);
      acc[0][2] = wmma_bf16(a0, b2, acc[0][2]);
      acc[0][3] = wmma_bf16(a0, b3, acc[0][3]);
      acc[1][0] = wmma_bf16(a1, b0, acc[1][0]);
      acc[1][1] = wmma_bf16(a1, b1, acc[1][1]);
      acc[1][2] = wmma_bf16(a1, b2, acc[1][2]);
      acc[1][3] = wmma_bf16(a1, b3, acc[1][3]);
    }
    __syncthreads();
  }

  // C layout: VGPR r, lanes 0-15 -> row r, lanes 16-31 -> row r+8; col = lane&15
  const int n    = lane & 15;
  const int half = lane >> 4;
#pragma unroll
  for (int i = 0; i < 2; ++i)
#pragma unroll
    for (int j = 0; j < 4; ++j)
#pragma unroll
      for (int r = 0; r < 8; ++r) {
        int row = m0 + wm * 32 + i * 16 + r + half * 8;
        C[(size_t)row * N + n0 + wn * 64 + j * 16 + n] = acc[i][j][r];
      }
}

// ---------------------------------------------------------------------------
// Kernel 3: RoPE + bf16 pack + layout change
//   q_f[b][t][h][d]  -> q_bf[b][h][t][d]    (rope)
//   k_f[b][t][kv][d] -> k_bf[b][kv][t][d]   (rope)
//   v_f[b][t][kv][d] -> vt_bf[b][kv][d][t]  (transposed for PV WMMA B-frags)
// ---------------------------------------------------------------------------
__global__ void __launch_bounds__(128) k_rope_pack(const float* __restrict__ q_f,
                                                   const float* __restrict__ k_f,
                                                   const float* __restrict__ v_f,
                                                   const float* __restrict__ cosT,
                                                   const float* __restrict__ sinT,
                                                   unsigned short* __restrict__ q_bf,
                                                   unsigned short* __restrict__ k_bf,
                                                   unsigned short* __restrict__ vt_bf) {
  const int bt = blockIdx.x;             // 0 .. B*T-1
  const int b  = bt >> 10;
  const int t  = bt & (TT - 1);
  const int d  = threadIdx.x;            // 0..127
  const float c = cosT[t * DD + d];
  const float s = sinT[t * DD + d];

#pragma unroll 4
  for (int h = 0; h < HH; ++h) {
    const float* x = q_f + ((size_t)bt * HH + h) * DD;
    float xp = (d < 64) ? -x[d + 64] : x[d - 64];
    float r  = x[d] * c + xp * s;
    q_bf[(((size_t)b * HH + h) * TT + t) * DD + d] = f32_to_bf16(r);
  }
#pragma unroll
  for (int h = 0; h < KVH; ++h) {
    const float* x = k_f + ((size_t)bt * KVH + h) * DD;
    float xp = (d < 64) ? -x[d + 64] : x[d - 64];
    float r  = x[d] * c + xp * s;
    k_bf[(((size_t)b * KVH + h) * TT + t) * DD + d] = f32_to_bf16(r);

    const float* xv = v_f + ((size_t)bt * KVH + h) * DD;
    vt_bf[(((size_t)b * KVH + h) * DD + d) * TT + t] = f32_to_bf16(xv[d]);
  }
}

// ---------------------------------------------------------------------------
// Kernel 4: flash attention (causal, GQA). One wave per (b, h, 16-query tile).
// Key step = 32 (two S fragments). Online softmax with wave32 half-shuffles.
// P is converted to bf16 through per-wave LDS to become a WMMA A-fragment.
// ---------------------------------------------------------------------------
__global__ void __launch_bounds__(256) k_flash_attn(const unsigned short* __restrict__ q_bf,
                                                    const unsigned short* __restrict__ k_bf,
                                                    const unsigned short* __restrict__ vt_bf,
                                                    unsigned short* __restrict__ o_bf) {
  __shared__ unsigned short ldsP[8][16][32];   // per-wave P staging (16 rows x 32 keys)

  const int w    = threadIdx.x >> 5;
  const int lane = threadIdx.x & 31;
  const int gw   = blockIdx.x * 8 + w;         // global wave id
  const int QT   = TT / 16;                    // 64 query tiles
  const int qt   = gw % QT;
  const int h    = (gw / QT) % HH;
  const int b    = gw / (QT * HH);
  const int kvh  = h >> 2;                     // Q_GROUP = 4
  const int q0   = qt * 16;

  const unsigned short* Q  = q_bf  + ((size_t)b * HH  + h)   * TT * DD;
  const unsigned short* Kp = k_bf  + ((size_t)b * KVH + kvh) * TT * DD;
  const unsigned short* Vt = vt_bf + ((size_t)b * KVH + kvh) * DD * TT;

  const int   n    = lane & 15;
  const int   half = lane >> 4;
  const float sc   = 0.08838834764831845f;     // 1/sqrt(128)

  // Q fragments for the whole 128-wide head dim (4 x 16x32)
  v16bf qf[4];
#pragma unroll
  for (int ds = 0; ds < 4; ++ds) qf[ds] = load_frag(Q, DD, q0, ds * 32, lane);

  v8f o[8];
#pragma unroll
  for (int dt = 0; dt < 8; ++dt) o[dt] = (v8f){0.f,0.f,0.f,0.f,0.f,0.f,0.f,0.f};
  float mrow[8], lrow[8];
#pragma unroll
  for (int r = 0; r < 8; ++r) { mrow[r] = -1e30f; lrow[r] = 0.f; }

  const int nks = (q0 + 16 + 31) / 32;         // causal: key steps needed
  for (int ks = 0; ks < nks; ++ks) {
    const int k0 = ks * 32;

    // ---- S = Q K^T (16 queries x 32 keys), 8 WMMAs over d=128 ----
    v8f s0 = {0.f,0.f,0.f,0.f,0.f,0.f,0.f,0.f};
    v8f s1 = {0.f,0.f,0.f,0.f,0.f,0.f,0.f,0.f};
#pragma unroll
    for (int ds = 0; ds < 4; ++ds) {
      v16bf bk0 = load_frag(Kp, DD, k0,      ds * 32, lane);
      v16bf bk1 = load_frag(Kp, DD, k0 + 16, ds * 32, lane);
      s0 = wmma_bf16(qf[ds], bk0, s0);
      s1 = wmma_bf16(qf[ds], bk1, s1);
    }

    // ---- scale + causal mask ----
#pragma unroll
    for (int r = 0; r < 8; ++r) {
      int qg = q0 + r + half * 8;
      float x0 = s0[r] * sc; if (k0 + n > qg)      x0 = NEGV;
      float x1 = s1[r] * sc; if (k0 + 16 + n > qg) x1 = NEGV;
      s0[r] = x0; s1[r] = x1;
    }

    // ---- online softmax: per-row reductions across the 16-lane half ----
#pragma unroll
    for (int r = 0; r < 8; ++r) {
      float mx = fmaxf(s0[r], s1[r]);
#pragma unroll
      for (int off = 8; off > 0; off >>= 1) mx = fmaxf(mx, __shfl_xor(mx, off, 32));
      float mnew = fmaxf(mrow[r], mx);
      float corr = __expf(mrow[r] - mnew);
      mrow[r] = mnew;
      float p0 = __expf(s0[r] - mnew);
      float p1 = __expf(s1[r] - mnew);
      s0[r] = p0; s1[r] = p1;
      float sum = p0 + p1;
#pragma unroll
      for (int off = 8; off > 0; off >>= 1) sum += __shfl_xor(sum, off, 32);
      lrow[r] = lrow[r] * corr + sum;
#pragma unroll
      for (int dt = 0; dt < 8; ++dt) o[dt][r] *= corr;   // rescale accumulator
    }

    // ---- P (C layout) -> LDS -> A-fragment layout ----
    unsigned short* pl = &ldsP[w][0][0];
#pragma unroll
    for (int r = 0; r < 8; ++r) {
      int row = r + half * 8;
      pl[row * 32 + n]      = f32_to_bf16(s0[r]);
      pl[row * 32 + 16 + n] = f32_to_bf16(s1[r]);
    }
    asm volatile("s_wait_dscnt 0" ::: "memory");   // wave-internal LDS RAW
    v16bf pf = load_frag(pl, 32, 0, 0, lane);

    // ---- O += P V : 8 accumulating WMMAs over the 128-wide head dim ----
#pragma unroll
    for (int dt = 0; dt < 8; ++dt) {
      v16bf vb = load_frag(Vt, TT, dt * 16, k0, lane);   // B[k=key][n=d] = Vt[d][key]
      o[dt] = wmma_bf16(pf, vb, o[dt]);
    }
  }

  // ---- epilogue: divide by row sums, pack bf16 into [token][4096] ----
#pragma unroll
  for (int dt = 0; dt < 8; ++dt) {
#pragma unroll
    for (int r = 0; r < 8; ++r) {
      float v = o[dt][r] / lrow[r];
      size_t tok = (size_t)b * TT + q0 + r + half * 8;
      o_bf[tok * KDIM + h * DD + dt * 16 + n] = f32_to_bf16(v);
    }
  }
}

// ---------------------------------------------------------------------------
// Host launcher
// ---------------------------------------------------------------------------
extern "C" void kernel_launch(void* const* d_in, const int* in_sizes, int n_in,
                              void* d_out, int out_size, void* d_ws, size_t ws_size,
                              hipStream_t stream) {
  const float* stm  = (const float*)d_in[0];
  const float* Wq   = (const float*)d_in[1];
  const float* Wk   = (const float*)d_in[2];
  const float* Wv   = (const float*)d_in[3];
  const float* Wo   = (const float*)d_in[4];
  const float* cosT = (const float*)d_in[5];
  const float* sinT = (const float*)d_in[6];
  float* out = (float*)d_out;

  char* ws = (char*)d_ws;
  size_t off = 0;
  auto alloc = [&](size_t bytes) -> char* {
    char* p = ws + off;
    off += (bytes + 255) & ~(size_t)255;
    return p;
  };

  const size_t nX  = (size_t)MDIM * KDIM;          // 2048*4096
  const size_t nWq = (size_t)KDIM * KDIM;
  const size_t nWk = (size_t)(KVH * DD) * KDIM;    // 1024*4096
  const size_t nKV = (size_t)MDIM * (KVH * DD);    // 2048*1024

  unsigned short* x_bf  = (unsigned short*)alloc(nX  * 2);
  unsigned short* wq_bf = (unsigned short*)alloc(nWq * 2);
  unsigned short* wk_bf = (unsigned short*)alloc(nWk * 2);
  unsigned short* wv_bf = (unsigned short*)alloc(nWk * 2);
  unsigned short* wo_bf = (unsigned short*)alloc(nWq * 2);
  float*          q_f   = (float*)alloc(nX  * 4);
  float*          k_f   = (float*)alloc(nKV * 4);
  float*          v_f   = (float*)alloc(nKV * 4);
  unsigned short* q_bf  = (unsigned short*)alloc(nX  * 2);
  unsigned short* k_bf  = (unsigned short*)alloc(nKV * 2);
  unsigned short* vt_bf = (unsigned short*)alloc(nKV * 2);
  unsigned short* o_bf  = (unsigned short*)alloc(nX  * 2);

  // 1) fp32 -> bf16 conversions
  k_f32_to_bf16<<<(int)(nX  / 1024), 256, 0, stream>>>(stm, x_bf,  (int)nX);
  k_f32_to_bf16<<<(int)(nWq / 1024), 256, 0, stream>>>(Wq,  wq_bf, (int)nWq);
  k_f32_to_bf16<<<(int)(nWk / 1024), 256, 0, stream>>>(Wk,  wk_bf, (int)nWk);
  k_f32_to_bf16<<<(int)(nWk / 1024), 256, 0, stream>>>(Wv,  wv_bf, (int)nWk);
  k_f32_to_bf16<<<(int)(nWq / 1024), 256, 0, stream>>>(Wo,  wo_bf, (int)nWq);

  // 2) QKV projections (WMMA GEMMs, fp32 out), block tile 128x128
  k_gemm_bf16<<<dim3(KDIM / 128, MDIM / 128), 256, 0, stream>>>(x_bf, wq_bf, q_f, MDIM, KDIM, KDIM);
  k_gemm_bf16<<<dim3((KVH * DD) / 128, MDIM / 128), 256, 0, stream>>>(x_bf, wk_bf, k_f, MDIM, KVH * DD, KDIM);
  k_gemm_bf16<<<dim3((KVH * DD) / 128, MDIM / 128), 256, 0, stream>>>(x_bf, wv_bf, v_f, MDIM, KVH * DD, KDIM);

  // 3) RoPE + bf16 pack + V transpose
  k_rope_pack<<<BB * TT, 128, 0, stream>>>(q_f, k_f, v_f, cosT, sinT, q_bf, k_bf, vt_bf);

  // 4) flash attention: B*H*(T/16) = 4096 waves -> 512 blocks of 8 waves
  k_flash_attn<<<(BB * HH * (TT / 16)) / 8, 256, 0, stream>>>(q_bf, k_bf, vt_bf, o_bf);

  // 5) output projection straight into d_out (fp32)
  k_gemm_bf16<<<dim3(KDIM / 128, MDIM / 128), 256, 0, stream>>>(o_bf, wo_bf, out, MDIM, KDIM, KDIM);
}
